// multiTimeAttention_43911745635031
// MI455X (gfx1250) — compile-verified, matrix operand order checked by reading, not run on hardware
//
#include <hip/hip_runtime.h>
#include <hip/hip_bf16.h>

typedef __attribute__((ext_vector_type(16))) _Float16 v16h;
typedef __attribute__((ext_vector_type(8)))  _Float16 v8h;
typedef __attribute__((ext_vector_type(8)))  float    v8f;

static constexpr int B_  = 8;
static constexpr int LQ_ = 96;
static constexpr int LK_ = 256;
static constexpr int DIM_ = 32;
static constexpr int E_  = 256;
static constexpr int H_  = 4;
static constexpr int NH_ = 128;
static constexpr int DK_ = 64;   // E/H

// ---------------- utility: wave32 reduction ----------------
__device__ __forceinline__ float wave_sum32(float v) {
#pragma unroll
    for (int off = 16; off > 0; off >>= 1)
        v += __shfl_xor(v, off, 32);
    return v;
}

// load one 16-half WMMA fragment: two contiguous 16B chunks (p[0..7], p[16..23])
__device__ __forceinline__ v16h load_frag(const _Float16* __restrict__ p) {
    v8h l = *(const v8h*)p;
    v8h h = *(const v8h*)(p + 16);
    return __builtin_shufflevector(l, h, 0, 1, 2, 3, 4, 5, 6, 7, 8, 9, 10, 11, 12, 13, 14, 15);
}

__device__ __forceinline__ v8f wmma_f16(v16h a, v16h b, v8f c) {
    return __builtin_amdgcn_wmma_f32_16x16x32_f16(false, a, false, b, (short)0, c, false, false);
}

// ---------------- f32 -> f16 convert ----------------
__global__ void cvt_f32_to_f16(const float* __restrict__ in, _Float16* __restrict__ out, int n) {
    int i = blockIdx.x * blockDim.x + threadIdx.x;
    if (i < n) out[i] = (_Float16)in[i];
}

// ---------------- f32 [K,N] -> f16 transposed [N,K] ----------------
__global__ void cvt_transpose_f32_to_f16(const float* __restrict__ in, _Float16* __restrict__ out,
                                         int K, int N) {
    int i = blockIdx.x * blockDim.x + threadIdx.x;
    if (i < K * N) {
        const int k = i / N, n = i % N;
        out[(size_t)n * K + k] = (_Float16)in[i];
    }
}

// ---------------- f16 WMMA GEMM with pre-transposed B: C[M,N] = A[M,K] @ Bt[N,K]^T ----------------
// one wave computes a 16x64 tile: A fragment reused across 4 independent WMMAs.
__global__ void gemm_wmma_f16_bt(const _Float16* __restrict__ A, const _Float16* __restrict__ Bt,
                                 float* __restrict__ C, int M, int N, int K) {
    const int wave = threadIdx.x >> 5;
    const int lane = threadIdx.x & 31;
    const int tile = blockIdx.x * (blockDim.x >> 5) + wave;
    const int tilesN = N >> 6;                 // groups of 64 columns
    const int numTiles = (M >> 4) * tilesN;
    if (tile >= numTiles) return;              // wave-uniform guard: EXEC all-1 for WMMA

    const int tm = tile / tilesN, tn = tile % tilesN;
    const int lo   = (lane >= 16) ? 8 : 0;
    const int m    = tm * 16 + (lane & 15);
    const int col  = tn * 64 + (lane & 15);    // this lane's column within each of 4 sub-tiles

    const _Float16* arow = A + (size_t)m * K;
    const _Float16* b0r  = Bt + (size_t)(col)      * K;
    const _Float16* b1r  = Bt + (size_t)(col + 16) * K;
    const _Float16* b2r  = Bt + (size_t)(col + 32) * K;
    const _Float16* b3r  = Bt + (size_t)(col + 48) * K;

    v8f acc0 = {}, acc1 = {}, acc2 = {}, acc3 = {};
    for (int kb = 0; kb < K; kb += 32) {
        __builtin_prefetch(arow + kb + 64, 0, 1);
        const v16h a  = load_frag(arow + kb + lo);
        const v16h f0 = load_frag(b0r + kb + lo);
        const v16h f1 = load_frag(b1r + kb + lo);
        const v16h f2 = load_frag(b2r + kb + lo);
        const v16h f3 = load_frag(b3r + kb + lo);
        acc0 = wmma_f16(a, f0, acc0);
        acc1 = wmma_f16(a, f1, acc1);
        acc2 = wmma_f16(a, f2, acc2);
        acc3 = wmma_f16(a, f3, acc3);
    }

    const int r0 = tm * 16 + lo;
    float* c0 = C + (size_t)r0 * N + col;
#pragma unroll
    for (int r = 0; r < 8; ++r) {
        c0[(size_t)r * N]      = acc0[r];
        c0[(size_t)r * N + 16] = acc1[r];
        c0[(size_t)r * N + 32] = acc2[r];
        c0[(size_t)r * N + 48] = acc3[r];
    }
}

// ---------------- row LayerNorm (+ optional bias), templated width & output type ----------------
template <int C, typename OUT>
__global__ void ln_rows(const float* __restrict__ in, const float* __restrict__ bias,
                        const float* __restrict__ g, const float* __restrict__ beta,
                        OUT* __restrict__ out, int rows) {
    constexpr int PL = C / 32;
    const int wave = threadIdx.x >> 5;
    const int lane = threadIdx.x & 31;
    const int row  = blockIdx.x * (blockDim.x >> 5) + wave;
    if (row >= rows) return;

    const float* p = in + (size_t)row * C;
    float v[PL];
    float s = 0.f;
#pragma unroll
    for (int i = 0; i < PL; ++i) {
        const int c = lane + 32 * i;
        const float bv = bias ? bias[c] : 0.f;
        v[i] = p[c] + bv;
        s += v[i];
    }
    s = wave_sum32(s);
    const float mean = s * (1.f / (float)C);
    float sq = 0.f;
#pragma unroll
    for (int i = 0; i < PL; ++i) { const float d = v[i] - mean; sq += d * d; }
    sq = wave_sum32(sq);
    const float rstd = rsqrtf(sq * (1.f / (float)C) + 1e-5f);
#pragma unroll
    for (int i = 0; i < PL; ++i) {
        const int c = lane + 32 * i;
        out[(size_t)row * C + c] = (OUT)((v[i] - mean) * rstd * g[c] + beta[c]);
    }
}

// ---------------- scores: per-head q @ k^T via WMMA ----------------
// one wave computes 16 queries x 64 keys (4 independent accumulators), K=DK=64.
// qn: f16 [B*LQ, E] row-major ; kn: f16 [B*LK, E] row-major ; scores: f32 [B,H,LQ,LK]
__global__ void scores_wmma(const _Float16* __restrict__ qn, const _Float16* __restrict__ kn,
                            float* __restrict__ scores) {
    constexpr int TQ = LQ_ / 16;   // 6
    constexpr int TG = LK_ / 64;   // 4 groups of 64 keys
    const int wave = threadIdx.x >> 5;
    const int lane = threadIdx.x & 31;
    const int tile = blockIdx.x * (blockDim.x >> 5) + wave;   // [0, B*H*TQ*TG)

    const int tg = tile % TG;
    int t = tile / TG;
    const int tq = t % TQ;  t /= TQ;
    const int h  = t % H_;
    const int b  = t / H_;

    const int lo   = (lane >= 16) ? 8 : 0;
    const int qrow = tq * 16 + (lane & 15);
    const int kp   = tg * 64 + (lane & 15);

    const _Float16* ap  = qn + ((size_t)(b * LQ_ + qrow)) * E_ + h * DK_;
    const _Float16* bp0 = kn + ((size_t)(b * LK_ + kp))      * E_ + h * DK_;
    const _Float16* bp1 = kn + ((size_t)(b * LK_ + kp + 16)) * E_ + h * DK_;
    const _Float16* bp2 = kn + ((size_t)(b * LK_ + kp + 32)) * E_ + h * DK_;
    const _Float16* bp3 = kn + ((size_t)(b * LK_ + kp + 48)) * E_ + h * DK_;

    v8f acc0 = {}, acc1 = {}, acc2 = {}, acc3 = {};
#pragma unroll
    for (int kb = 0; kb < DK_; kb += 32) {
        const v16h a  = load_frag(ap  + kb + lo);
        const v16h f0 = load_frag(bp0 + kb + lo);
        const v16h f1 = load_frag(bp1 + kb + lo);
        const v16h f2 = load_frag(bp2 + kb + lo);
        const v16h f3 = load_frag(bp3 + kb + lo);
        acc0 = wmma_f16(a, f0, acc0);
        acc1 = wmma_f16(a, f1, acc1);
        acc2 = wmma_f16(a, f2, acc2);
        acc3 = wmma_f16(a, f3, acc3);
    }

    const float scale = 0.125f;               // 1/sqrt(DK=64)
    const int q0 = tq * 16 + lo;
    const int c  = tg * 64 + (lane & 15);
    float* outp = scores + ((size_t)(b * H_ + h) * LQ_ + q0) * LK_ + c;
#pragma unroll
    for (int r = 0; r < 8; ++r) {
        outp[(size_t)r * LK_]      = acc0[r] * scale;
        outp[(size_t)r * LK_ + 16] = acc1[r] * scale;
        outp[(size_t)r * LK_ + 32] = acc2[r] * scale;
        outp[(size_t)r * LK_ + 48] = acc3[r] * scale;
    }
}

// ---------------- per-feature masked online softmax + AV ----------------
// one wave per (b,h,q); lane == feature d (DIM==32==wave32). Writes x[B,LQ,H*DIM].
__global__ void attn_softmax_av(const float* __restrict__ scores, const float* __restrict__ value,
                                const int* __restrict__ mask, float* __restrict__ x) {
    const int wave = threadIdx.x >> 5;
    const int lane = threadIdx.x & 31;                 // d
    const int idx  = blockIdx.x * (blockDim.x >> 5) + wave;   // (b*H + h)*LQ + q
    if (idx >= B_ * H_ * LQ_) return;

    const int q = idx % LQ_;
    int t = idx / LQ_;
    const int h = t % H_;
    const int b = t / H_;

    const float* sp = scores + (size_t)idx * LK_;
    const float* vp = value + ((size_t)b * LK_) * DIM_ + lane;
    const int*   mp = mask  + ((size_t)b * LK_) * DIM_ + lane;

    float m = -INFINITY, den = 0.f, acc = 0.f;
    for (int k = 0; k < LK_; ++k) {
        const float s  = sp[k];
        const float xk = (mp[(size_t)k * DIM_] == 0) ? -1e9f : s;
        const float nm = fmaxf(m, xk);
        const float corr = __expf(m - nm);
        const float e    = __expf(xk - nm);
        den = den * corr + e;
        acc = acc * corr + e * vp[(size_t)k * DIM_];
        m = nm;
    }
    x[((size_t)(b * LQ_ + q)) * (H_ * DIM_) + h * DIM_ + lane] = acc / den;
}

// ---------------- host side ----------------
extern "C" void kernel_launch(void* const* d_in, const int* in_sizes, int n_in,
                              void* d_out, int out_size, void* d_ws, size_t ws_size,
                              hipStream_t stream) {
    (void)in_sizes; (void)n_in; (void)out_size; (void)ws_size;

    const float* query = (const float*)d_in[0];   // [8,96,256]
    const float* key   = (const float*)d_in[1];   // [8,256,256]
    const float* value = (const float*)d_in[2];   // [8,256,32]
    const int*   mask  = (const int*)  d_in[3];   // [8,256,32]
    const float* Wq    = (const float*)d_in[4];
    const float* bq    = (const float*)d_in[5];
    const float* gq    = (const float*)d_in[6];
    const float* betaq = (const float*)d_in[7];
    const float* Wk    = (const float*)d_in[8];
    const float* bk    = (const float*)d_in[9];
    const float* gk    = (const float*)d_in[10];
    const float* betak = (const float*)d_in[11];
    const float* ga    = (const float*)d_in[12];
    const float* beta_a= (const float*)d_in[13];
    const float* Wo    = (const float*)d_in[14];  // [128,128]
    const float* bo    = (const float*)d_in[15];
    const float* go    = (const float*)d_in[16];
    const float* betao = (const float*)d_in[17];
    float* out = (float*)d_out;                   // [8,96,128]

    const int nQ  = B_ * LQ_ * E_;     // 196608
    const int nK  = B_ * LK_ * E_;     // 524288
    const int nWq = E_ * E_;           // 65536
    const int nWo = (H_ * DIM_) * NH_; // 16384
    const int MQ = B_ * LQ_;           // 768 rows
    const int MK = B_ * LK_;           // 2048 rows
    const int nScores = B_ * H_ * LQ_ * LK_;  // 786432
    const int nX = MQ * (H_ * DIM_);   // 98304

    char* w = (char*)d_ws;
    size_t off = 0;
    auto alloc = [&](size_t bytes) -> void* {
        void* p = w + off;
        off = (off + bytes + 255) & ~(size_t)255;
        return p;
    };
    _Float16* qf16   = (_Float16*)alloc((size_t)nQ * 2);
    _Float16* kf16   = (_Float16*)alloc((size_t)nK * 2);
    _Float16* WqT    = (_Float16*)alloc((size_t)nWq * 2);   // [N=E, K=E]
    _Float16* WkT    = (_Float16*)alloc((size_t)nWq * 2);
    _Float16* WoT    = (_Float16*)alloc((size_t)nWo * 2);   // [N=NH, K=128]
    float*    qproj  = (float*)   alloc((size_t)nQ * 4);
    float*    kproj  = (float*)   alloc((size_t)nK * 4);
    _Float16* qn     = (_Float16*)alloc((size_t)nQ * 2);
    _Float16* kn     = (_Float16*)alloc((size_t)nK * 2);
    float*    scores = (float*)   alloc((size_t)nScores * 4);
    float*    attx   = (float*)   alloc((size_t)nX * 4);
    _Float16* xn     = (_Float16*)alloc((size_t)nX * 2);
    float*    y      = (float*)   alloc((size_t)nX * 4);

    // 1) converts to f16 (weights transposed to [N,K] so WMMA B-fragments are contiguous)
    cvt_f32_to_f16<<<(nQ  + 255) / 256, 256, 0, stream>>>(query, qf16, nQ);
    cvt_f32_to_f16<<<(nK  + 255) / 256, 256, 0, stream>>>(key,   kf16, nK);
    cvt_transpose_f32_to_f16<<<(nWq + 255) / 256, 256, 0, stream>>>(Wq, WqT, E_, E_);
    cvt_transpose_f32_to_f16<<<(nWq + 255) / 256, 256, 0, stream>>>(Wk, WkT, E_, E_);
    cvt_transpose_f32_to_f16<<<(nWo + 255) / 256, 256, 0, stream>>>(Wo, WoT, H_ * DIM_, NH_);

    // 2) projections via WMMA: 16x64 tile per wave
    {
        const int tiles = (MQ / 16) * (E_ / 64);        // 192
        gemm_wmma_f16_bt<<<tiles / 4, 128, 0, stream>>>(qf16, WqT, qproj, MQ, E_, E_);
    }
    {
        const int tiles = (MK / 16) * (E_ / 64);        // 512
        gemm_wmma_f16_bt<<<tiles / 4, 128, 0, stream>>>(kf16, WkT, kproj, MK, E_, E_);
    }

    // 3) LN(+bias) -> f16
    ln_rows<E_, _Float16><<<MQ / 8, 256, 0, stream>>>(qproj, bq, gq, betaq, qn, MQ);
    ln_rows<E_, _Float16><<<MK / 8, 256, 0, stream>>>(kproj, bk, gk, betak, kn, MK);

    // 4) scores via WMMA (per head, 16 queries x 64 keys per wave)
    {
        const int tiles = B_ * H_ * (LQ_ / 16) * (LK_ / 64);  // 768
        scores_wmma<<<tiles / 4, 128, 0, stream>>>(qn, kn, scores);
    }

    // 5) masked per-feature online softmax + AV; lanes == DIM features
    {
        const int waves = B_ * H_ * LQ_;                      // 3072
        attn_softmax_av<<<waves / 4, 128, 0, stream>>>(scores, value, mask, attx);
    }

    // 6) LN (no bias) -> f16, out-proj WMMA, final LN(+bo) -> f32 out
    ln_rows<NH_, _Float16><<<MQ / 8, 256, 0, stream>>>(attx, nullptr, ga, beta_a, xn, MQ);
    {
        const int tiles = (MQ / 16) * (NH_ / 64);             // 96
        gemm_wmma_f16_bt<<<tiles / 4, 128, 0, stream>>>(xn, WoT, y, MQ, NH_, NH_);
    }
    ln_rows<NH_, float><<<MQ / 8, 256, 0, stream>>>(y, bo, go, betao, out, MQ);
}
